// BlockCorrelation_87290915323955
// MI455X (gfx1250) — compile-verified, matrix operand order checked by reading, not run on hardware
//
#include <hip/hip_runtime.h>
#include <hip/hip_bf16.h>
#include <math.h>

typedef __attribute__((ext_vector_type(16))) _Float16 v16h;
typedef __attribute__((ext_vector_type(8)))  _Float16 v8h;
typedef __attribute__((ext_vector_type(8)))  float    v8f;

#define DIM 1024
#define NTOK 2048
#define NHEADS 8
#define HDIM 128
#define NGROUPS 32
#define HW 64

// ---------------------------------------------------------------------------
// WMMA fragment loader. 16-bit A-matrix layout (CDNA5 ISA 7.12.2):
//   lanes 0-15 : M = lane,    VGPR0-3 = K 0..7,  VGPR4-7 = K 16..23
//   lanes 16-31: M = lane-16, VGPR0-3 = K 8..15, VGPR4-7 = K 24..31
// => per lane: two contiguous 16-byte loads from a row-major [M][K] tile.
// B fragments (32x16, K x N) use the mirrored layout with lane = column, so
// the same loader works on a row-major [N][K] tile (i.e. W stored as rows).
// ---------------------------------------------------------------------------
__device__ __forceinline__ v16h ldfrag(const _Float16* base, int ld) {
  const int lane = threadIdx.x & 31;
  const _Float16* p = base + (size_t)(lane & 15) * ld + ((lane >> 4) << 3);
  v8h lo = *(const v8h*)(p);
  v8h hi = *(const v8h*)(p + 16);
  v16h r;
#pragma unroll
  for (int i = 0; i < 8; ++i) { r[i] = lo[i]; r[8 + i] = hi[i]; }
  return r;
}

__device__ __forceinline__ v8f wmma_f16(v16h a, v16h b, v8f c) {
  return __builtin_amdgcn_wmma_f32_16x16x32_f16(false, a, false, b, (short)0, c,
                                                false, false);
}

// ---------------------------------------------------------------------------
// 1) feat = mean over H*W.  One wave32 per (n,c) row of 64 contiguous floats.
// ---------------------------------------------------------------------------
__global__ void mean_kernel(const float* __restrict__ x, float* __restrict__ feat) {
  const int row  = blockIdx.x * 8 + (threadIdx.x >> 5);
  const int lane = threadIdx.x & 31;
  const float2* p = (const float2*)(x + (size_t)row * HW);
  float2 v = p[lane];
  float s = v.x + v.y;
#pragma unroll
  for (int m = 1; m < 32; m <<= 1) s += __shfl_xor(s, m, 32);
  if (lane == 0) feat[row] = s * (1.0f / (float)HW);
}

// ---------------------------------------------------------------------------
// 2) LayerNorm over C=1024; emit f16.  One 256-thread block per token.
// ---------------------------------------------------------------------------
__global__ void ln_kernel(const float* __restrict__ feat,
                          const float* __restrict__ w, const float* __restrict__ b,
                          _Float16* __restrict__ featn) {
  const int n = blockIdx.x, t = threadIdx.x;
  __shared__ float sh[256], sh2[256];
  float vals[4];
  float s = 0.f, s2 = 0.f;
#pragma unroll
  for (int i = 0; i < 4; ++i) {
    float v = feat[(size_t)n * DIM + t + i * 256];
    vals[i] = v; s += v; s2 += v * v;
  }
  sh[t] = s; sh2[t] = s2;
  __syncthreads();
  for (int off = 128; off > 0; off >>= 1) {
    if (t < off) { sh[t] += sh[t + off]; sh2[t] += sh2[t + off]; }
    __syncthreads();
  }
  const float mu  = sh[0] * (1.0f / DIM);
  const float var = sh2[0] * (1.0f / DIM) - mu * mu;
  const float rs  = rsqrtf(var + 1e-5f);
#pragma unroll
  for (int i = 0; i < 4; ++i) {
    int c = t + i * 256;
    featn[(size_t)n * DIM + c] = (_Float16)((vals[i] - mu) * rs * w[c] + b[c]);
  }
}

// ---------------------------------------------------------------------------
// f32 -> f16 convert (weights)
// ---------------------------------------------------------------------------
__global__ void cvt_kernel(const float* __restrict__ in, _Float16* __restrict__ out,
                           int n) {
  int i = blockIdx.x * 256 + threadIdx.x;
  if (i < n) out[i] = (_Float16)in[i];
}

// ---------------------------------------------------------------------------
// 3/5) GEMM:  out[m][n] = sum_k A[m][k] * B[n][k] + bias[n]
//   A row-major [M][K] f16, B row-major [N][K] f16 (i.e. W, used as B^T).
//   Block = 8 waves, wave tile 32x32 (2x2 fragments), block tile 128x64.
//   mode 0: QKV epilogue (q,k row-major f16; v stored transposed per head)
//   mode 1: plain f32 output
// ---------------------------------------------------------------------------
__global__ void gemm_kernel(const _Float16* __restrict__ A,
                            const _Float16* __restrict__ B,
                            const float* __restrict__ bias,
                            int K, int mode,
                            _Float16* __restrict__ qout, _Float16* __restrict__ kout,
                            _Float16* __restrict__ vT, float* __restrict__ fout,
                            int Ntot) {
  const int w    = threadIdx.x >> 5;
  const int lane = threadIdx.x & 31;
  const int m0 = blockIdx.y * 128 + (w >> 1) * 32;
  const int n0 = blockIdx.x * 64  + (w & 1) * 32;

  v8f acc[2][2];
#pragma unroll
  for (int i = 0; i < 2; ++i)
#pragma unroll
    for (int j = 0; j < 2; ++j) acc[i][j] = (v8f){};

  for (int kk = 0; kk < K; kk += 32) {
    v16h a0 = ldfrag(A + (size_t)m0 * K + kk, K);
    v16h a1 = ldfrag(A + (size_t)(m0 + 16) * K + kk, K);
    v16h b0 = ldfrag(B + (size_t)n0 * K + kk, K);
    v16h b1 = ldfrag(B + (size_t)(n0 + 16) * K + kk, K);
    acc[0][0] = wmma_f16(a0, b0, acc[0][0]);
    acc[0][1] = wmma_f16(a0, b1, acc[0][1]);
    acc[1][0] = wmma_f16(a1, b0, acc[1][0]);
    acc[1][1] = wmma_f16(a1, b1, acc[1][1]);
  }

  const int hi = lane >> 4, nl = lane & 15;
#pragma unroll
  for (int i = 0; i < 2; ++i)
#pragma unroll
    for (int j = 0; j < 2; ++j)
#pragma unroll
      for (int r = 0; r < 8; ++r) {
        const int m = m0 + i * 16 + r + hi * 8;
        const int n = n0 + j * 16 + nl;
        const float val = acc[i][j][r] + bias[n];
        if (mode == 0) {
          if (n < DIM) {
            qout[(size_t)m * DIM + n] = (_Float16)val;
          } else if (n < 2 * DIM) {
            kout[(size_t)m * DIM + (n - DIM)] = (_Float16)val;
          } else {
            const int nn = n - 2 * DIM;
            const int hh = nn >> 7, d = nn & 127;
            vT[((size_t)hh * HDIM + d) * NTOK + m] = (_Float16)val;  // [h][d][n]
          }
        } else {
          fout[(size_t)m * Ntot + n] = val;
        }
      }
}

// ---------------------------------------------------------------------------
// 4) Masked flash attention. One wave = 16 queries of one head over all keys.
//    Block = 8 waves (128 queries). grid = (2048/128, 8 heads).
// ---------------------------------------------------------------------------
__global__ void attn_kernel(const _Float16* __restrict__ qb,
                            const _Float16* __restrict__ kb,
                            const _Float16* __restrict__ vT,
                            const int* __restrict__ bi,
                            _Float16* __restrict__ attn_out) {
  __shared__ __align__(16) _Float16 pbuf[8 * 16 * 32];  // per-wave 16x32 P tile
  const int w    = threadIdx.x >> 5;
  const int lane = threadIdx.x & 31;
  const int h    = blockIdx.y;
  const int q0   = blockIdx.x * 128 + w * 16;
  const int hi = lane >> 4, nl = lane & 15;
  _Float16* pw = pbuf + w * 512;

  // Q fragments: 16 x 128 = 4 K-slices of 32
  v16h qf[4];
#pragma unroll
  for (int kk = 0; kk < 4; ++kk)
    qf[kk] = ldfrag(qb + (size_t)q0 * DIM + h * HDIM + kk * 32, DIM);

  int qbi[8];
#pragma unroll
  for (int r = 0; r < 8; ++r) qbi[r] = bi[q0 + r + hi * 8];

  float mrun[8], lrun[8];
  v8f acc[8];
#pragma unroll
  for (int r = 0; r < 8; ++r) { mrun[r] = -1.0e30f; lrun[r] = 0.f; }
#pragma unroll
  for (int dt = 0; dt < 8; ++dt) acc[dt] = (v8f){};

  const float scale = 0.08838834764831845f;  // 1/sqrt(128)

  for (int j0 = 0; j0 < NTOK; j0 += 32) {
    // S = Q @ K^T : two 16x16 fragments (keys j0..+15, j0+16..+31)
    v8f s0 = (v8f){}, s1 = (v8f){};
#pragma unroll
    for (int kk = 0; kk < 4; ++kk) {
      v16h kf0 = ldfrag(kb + (size_t)j0 * DIM + h * HDIM + kk * 32, DIM);
      v16h kf1 = ldfrag(kb + (size_t)(j0 + 16) * DIM + h * HDIM + kk * 32, DIM);
      s0 = wmma_f16(qf[kk], kf0, s0);
      s1 = wmma_f16(qf[kk], kf1, s1);
    }

    const int kb0 = bi[j0 + nl];
    const int kb1 = bi[j0 + 16 + nl];

    float pr0[8], pr1[8];
#pragma unroll
    for (int r = 0; r < 8; ++r) {
      float a  = s0[r] * scale;
      float b2 = s1[r] * scale;
      if (qbi[r] != kb0) a  = -3.0e38f;
      if (qbi[r] != kb1) b2 = -3.0e38f;
      float t = fmaxf(a, b2);
#pragma unroll
      for (int m = 1; m < 16; m <<= 1) t = fmaxf(t, __shfl_xor(t, m, 16));
      const float mn = fmaxf(mrun[r], t);
      const float sc = __expf(mrun[r] - mn);
      const float e0 = __expf(a - mn);
      const float e1 = __expf(b2 - mn);
      float rs = e0 + e1;
#pragma unroll
      for (int m = 1; m < 16; m <<= 1) rs += __shfl_xor(rs, m, 16);
      lrun[r] = lrun[r] * sc + rs;
      mrun[r] = mn;
#pragma unroll
      for (int dt = 0; dt < 8; ++dt) acc[dt][r] *= sc;
      pr0[r] = e0; pr1[r] = e1;
    }

    // C-layout f32 -> A-layout f16 for P via LDS (in-order per-wave DS ops)
#pragma unroll
    for (int r = 0; r < 8; ++r) {
      const int mloc = r + hi * 8;
      pw[mloc * 32 + nl]      = (_Float16)pr0[r];
      pw[mloc * 32 + 16 + nl] = (_Float16)pr1[r];
    }
    v16h pA = ldfrag(pw, 32);

    // O += P @ V  (V^T stored [h][d][n] so B fragments are contiguous)
#pragma unroll
    for (int dt = 0; dt < 8; ++dt) {
      v16h vf = ldfrag(vT + ((size_t)h * HDIM + dt * 16) * NTOK + j0, NTOK);
      acc[dt] = wmma_f16(pA, vf, acc[dt]);
    }
  }

#pragma unroll
  for (int dt = 0; dt < 8; ++dt)
#pragma unroll
    for (int r = 0; r < 8; ++r) {
      const int m = r + hi * 8;
      const float o = acc[dt][r] / lrun[r];
      attn_out[(size_t)(q0 + m) * DIM + h * HDIM + dt * 16 + nl] = (_Float16)o;
    }
}

// ---------------------------------------------------------------------------
// 7) group counts
// ---------------------------------------------------------------------------
__global__ void counts_kernel(const int* __restrict__ bi, int* __restrict__ counts) {
  __shared__ int c[NGROUPS];
  if (threadIdx.x < NGROUPS) c[threadIdx.x] = 0;
  __syncthreads();
  for (int i = threadIdx.x; i < NTOK; i += 256) atomicAdd(&c[bi[i] & 31], 1);
  __syncthreads();
  if (threadIdx.x < NGROUPS) counts[threadIdx.x] = c[threadIdx.x];
}

// ---------------------------------------------------------------------------
// 8) out = x + where(valid, gamma*delta, 0), delta broadcast over H,W.
//    float4 vectorized: one thread per 4 of the 64 HW elements.
// ---------------------------------------------------------------------------
__global__ void final_kernel(const float* __restrict__ x,
                             const float* __restrict__ proj,
                             const int* __restrict__ bi,
                             const int* __restrict__ counts,
                             const float* __restrict__ gamma,
                             float* __restrict__ out) {
  const size_t gid = (size_t)blockIdx.x * 256 + threadIdx.x;  // N*C*16 total
  const int n = (int)(gid >> 14);
  const int c = (int)((gid >> 4) & 1023);
  const bool valid = counts[bi[n] & 31] > 1;
  const float d = valid ? gamma[0] * proj[(size_t)n * DIM + c] : 0.f;
  float4 v = ((const float4*)x)[gid];
  v.x += d; v.y += d; v.z += d; v.w += d;
  ((float4*)out)[gid] = v;
}

// ---------------------------------------------------------------------------
extern "C" void kernel_launch(void* const* d_in, const int* in_sizes, int n_in,
                              void* d_out, int out_size, void* d_ws, size_t ws_size,
                              hipStream_t stream) {
  (void)in_sizes; (void)n_in; (void)out_size; (void)ws_size;
  const float* x          = (const float*)d_in[0];
  const int*   bidx       = (const int*)d_in[1];
  const float* ln_w       = (const float*)d_in[2];
  const float* ln_b       = (const float*)d_in[3];
  const float* in_proj_w  = (const float*)d_in[4];
  const float* in_proj_b  = (const float*)d_in[5];
  const float* out_proj_w = (const float*)d_in[6];
  const float* out_proj_b = (const float*)d_in[7];
  const float* gamma      = (const float*)d_in[8];
  float* out = (float*)d_out;

  char* ws = (char*)d_ws;
  size_t off = 0;
  auto carve = [&](size_t bytes) { void* p = ws + off; off = (off + bytes + 255) & ~(size_t)255; return p; };
  float*    feat  = (float*)   carve((size_t)NTOK * DIM * 4);
  _Float16* featn = (_Float16*)carve((size_t)NTOK * DIM * 2);
  _Float16* wqkv  = (_Float16*)carve((size_t)3 * DIM * DIM * 2);
  _Float16* wout  = (_Float16*)carve((size_t)DIM * DIM * 2);
  _Float16* qbuf  = (_Float16*)carve((size_t)NTOK * DIM * 2);
  _Float16* kbuf  = (_Float16*)carve((size_t)NTOK * DIM * 2);
  _Float16* vT    = (_Float16*)carve((size_t)NTOK * DIM * 2);
  _Float16* attn  = (_Float16*)carve((size_t)NTOK * DIM * 2);
  float*    proj  = (float*)   carve((size_t)NTOK * DIM * 4);
  int*      cnts  = (int*)     carve(NGROUPS * 4);

  // 1) spatial mean: N*C rows, 8 rows per block
  mean_kernel<<<(NTOK * DIM) / 8, 256, 0, stream>>>(x, feat);
  // 2) layernorm + f16 cast
  ln_kernel<<<NTOK, 256, 0, stream>>>(feat, ln_w, ln_b, featn);
  // 3) weight conversion
  cvt_kernel<<<(3 * DIM * DIM + 255) / 256, 256, 0, stream>>>(in_proj_w, wqkv, 3 * DIM * DIM);
  cvt_kernel<<<(DIM * DIM + 255) / 256, 256, 0, stream>>>(out_proj_w, wout, DIM * DIM);
  // 4) QKV GEMM (M=2048, N=3072, K=1024), q/k/vT epilogue
  gemm_kernel<<<dim3(3 * DIM / 64, NTOK / 128), 256, 0, stream>>>(
      featn, wqkv, in_proj_b, DIM, 0, qbuf, kbuf, vT, nullptr, 3 * DIM);
  // 5) masked flash attention
  attn_kernel<<<dim3(NTOK / 128, NHEADS), 256, 0, stream>>>(qbuf, kbuf, vT, bidx, attn);
  // 6) out_proj GEMM (M=2048, N=1024, K=1024) -> f32
  gemm_kernel<<<dim3(DIM / 64, NTOK / 128), 256, 0, stream>>>(
      attn, wout, out_proj_b, DIM, 1, nullptr, nullptr, nullptr, proj, DIM);
  // 7) group counts
  counts_kernel<<<1, 256, 0, stream>>>(bidx, cnts);
  // 8) residual broadcast add
  final_kernel<<<(NTOK * DIM * 16) / 256, 256, 0, stream>>>(x, proj, bidx, cnts, gamma, out);
}
